// _RPN_9225589751818
// MI455X (gfx1250) — matrix operand order repaired and checked by reading, not compiled, not run on hardware
//
#include <hip/hip_runtime.h>
#include <hip/hip_bf16.h>

#define NB        4
#define CIN       1024
#define HH        64
#define WWD       64
#define HWSZ      (HH*WWD)        // 4096
#define MROWS     (NB*HWSZ)       // 16384
#define CMID      512
#define KCONV     (CIN*9)         // 9216
#define NHEAD     96              // 18 cls + 18 atten + 36 bbox + 20 proto + 4 pad
#define NANCH     (HWSZ*9)        // 36864
#define PRE       6000
#define POST      300
#define NBIN      8192

typedef unsigned short bfu16;
typedef __attribute__((ext_vector_type(16))) __bf16 v16bf;
typedef __attribute__((ext_vector_type(8)))  float  v8f;

union Frag16 { uint4 u[2]; v16bf v; };

__device__ __forceinline__ bfu16 f2bf(float f) {
  unsigned int u = __float_as_uint(f);
  u += 0x7FFFu + ((u >> 16) & 1u);
  return (bfu16)(u >> 16);
}

__device__ __forceinline__ uint4 and4(uint4 a, unsigned m) {
  a.x &= m; a.y &= m; a.z &= m; a.w &= m; return a;
}

// ---------------- small prep kernels ----------------

__global__ void k_zero_int(int* p, int n) {
  int i = blockIdx.x * blockDim.x + threadIdx.x;
  if (i < n) p[i] = 0;
}

// NCHW f32 -> NHWC bf16
__global__ void k_cvt_x(const float* __restrict__ x, bfu16* __restrict__ out) {
  int i = blockIdx.x * blockDim.x + threadIdx.x;
  if (i >= MROWS * CIN) return;
  int c  = i & (CIN - 1);
  int n  = i >> 10;              // b*HW + h*W + w
  int hw = n & (HWSZ - 1);
  int b  = n >> 12;
  out[i] = f2bf(x[((size_t)(b * CIN + c)) * HWSZ + hw]);
}

// conv_w (O,CI,3,3) f32 -> Wg[o][tap*1024+ci] bf16 (K contiguous, tap-major)
__global__ void k_cvt_wconv(const float* __restrict__ w, bfu16* __restrict__ out) {
  int i = blockIdx.x * blockDim.x + threadIdx.x;
  if (i >= CMID * KCONV) return;
  int o   = i / KCONV;
  int t   = i - o * KCONV;
  int tap = t >> 10;
  int ci  = t & 1023;
  out[i] = f2bf(w[((size_t)(o * CIN + ci)) * 9 + tap]);
}

// gang_w1 (512,518) -> first 512 cols as bf16 [512][512]
__global__ void k_cvt_w1g(const float* __restrict__ w, bfu16* __restrict__ out) {
  int i = blockIdx.x * blockDim.x + threadIdx.x;
  if (i >= CMID * CMID) return;
  int o = i >> 9, k = i & 511;
  out[i] = f2bf(w[(size_t)o * 518 + k]);
}

// P_ep = mean(P_ang, axis 0); att_sum = sum(attentions[1:], axis 0)
__global__ void k_pep_attsum(const float* __restrict__ P_ang, const float* __restrict__ att,
                             float* __restrict__ pep, float* __restrict__ attsum) {
  int t = threadIdx.x;
  for (int d = t; d < 2048; d += blockDim.x) {
    float s = 0.f;
    for (int r = 1; r < 20; ++r) s += att[(size_t)r * 2048 + d];
    attsum[d] = s;
  }
  if (t < 6) {
    float s = 0.f;
    for (int r = 0; r < 20; ++r) s += P_ang[r * 6 + t];
    pep[t] = s * (1.0f / 20.0f);
  }
}

// gate_bias[o] = gang_b1[o] + sum_j gang_w1[o][512+j]*P_ep[j]
__global__ void k_gate_bias(const float* __restrict__ gw1, const float* __restrict__ gb1,
                            const float* __restrict__ pep, float* __restrict__ gbias) {
  int o = blockIdx.x * blockDim.x + threadIdx.x;
  if (o >= CMID) return;
  float s = gb1[o];
  for (int j = 0; j < 6; ++j) s += gw1[(size_t)o * 518 + 512 + j] * pep[j];
  gbias[o] = s;
}

// generic dense vec@matT: out[o] = act(dot(in, W[o]) + b[o]); act: 0 none, 1 relu, 2 sigmoid
__global__ void k_vecmat(const float* __restrict__ in, const float* __restrict__ W,
                         const float* __restrict__ b, float* __restrict__ out,
                         int Nout, int Kin, int act) {
  int o = blockIdx.x * blockDim.x + threadIdx.x;
  if (o >= Nout) return;
  const float* wr = W + (size_t)o * Kin;
  float s = 0.f;
  for (int k = 0; k < Kin; ++k) s += in[k] * wr[k];
  s += b[o];
  if (act == 1) s = fmaxf(s, 0.f);
  else if (act == 2) s = 1.f / (1.f + expf(-s));
  out[o] = s;
}

// dyn_w row block of Whead (rows 18..35): whead[18*512 + i] = bf16(dot(h2, meta_w3[i]) + b3[i])
__global__ void k_dyn(const float* __restrict__ h2, const float* __restrict__ w3,
                      const float* __restrict__ b3, bfu16* __restrict__ whead) {
  int i = blockIdx.x * blockDim.x + threadIdx.x;
  if (i >= 18 * CMID) return;
  const float* wr = w3 + (size_t)i * 128;
  float s = b3[i];
  for (int k = 0; k < 128; ++k) s += h2[k] * wr[k];
  whead[18 * CMID + i] = f2bf(s);
}

// cproto[j] = dot(proto_b, P_pure[j])
__global__ void k_cproto(const float* __restrict__ pb, const float* __restrict__ Pp,
                         float* __restrict__ cp) {
  int j = threadIdx.x;
  if (j >= 20) return;
  const float* pr = Pp + (size_t)j * 2048;
  float s = 0.f;
  for (int d = 0; d < 2048; ++d) s += pb[d] * pr[d];
  cp[j] = s;
}

// W2 rows (72..91): whead[(72+j)*512+k] = bf16(sum_d proto_w[d][k]*P_pure[j][d])
__global__ void k_w2proto(const float* __restrict__ pw, const float* __restrict__ Pp,
                          bfu16* __restrict__ whead) {
  int i = blockIdx.x * blockDim.x + threadIdx.x;
  if (i >= 20 * CMID) return;
  int j = i >> 9, k = i & 511;
  const float* pr = Pp + (size_t)j * 2048;
  float s = 0.f;
  for (int d = 0; d < 2048; ++d) s += pw[(size_t)d * CMID + k] * pr[d];
  whead[(size_t)(72 + j) * CMID + k] = f2bf(s);
}

// static rows of Whead + head bias vector
__global__ void k_whead_static(const float* __restrict__ cls_w, const float* __restrict__ bbox_w,
                               const float* __restrict__ cls_b, const float* __restrict__ bbox_b,
                               const float* __restrict__ cproto,
                               bfu16* __restrict__ whead, float* __restrict__ biash) {
  int i = blockIdx.x * blockDim.x + threadIdx.x;
  if (i < NHEAD * CMID) {
    int row = i >> 9, k = i & 511;
    if (row < 18)                     whead[i] = f2bf(cls_w[(size_t)row * CMID + k]);
    else if (row >= 36 && row < 72)   whead[i] = f2bf(bbox_w[(size_t)(row - 36) * CMID + k]);
    else if (row >= 92)               whead[i] = 0;
    // rows 18..35 (dyn) and 72..91 (proto) filled elsewhere
  }
  if (i < NHEAD) {
    float v = 0.f;
    if (i < 18)                 v = cls_b[i];
    else if (i >= 36 && i < 72) v = bbox_b[i - 36];
    else if (i >= 72 && i < 92) v = cproto[i - 72];
    biash[i] = v;
  }
}

// ---------------- WMMA GEMMs ----------------

// implicit-GEMM 3x3 conv: one wave computes a 16(M) x 64(N) strip (4 accumulators
// sharing one A fragment).  All 4 B fragments are staged before the WMMA burst so
// the load->use waits are partial instead of s_wait_loadcnt 0.
__global__ void conv3_wmma(const bfu16* __restrict__ X, const bfu16* __restrict__ Wg,
                           const float* __restrict__ bias,
                           float* __restrict__ RC, bfu16* __restrict__ RCb) {
  int lane = threadIdx.x & 31;
  int half = lane >> 4, l15 = lane & 15;
  int n0 = blockIdx.x * 64;                 // output-channel strip (4 x 16)
  int m0 = blockIdx.y * 16;                 // spatial tile: 16 consecutive w, same h
  int b  = m0 >> 12;
  int hw = m0 & (HWSZ - 1);
  int h  = hw >> 6, w0 = hw & 63;
  int wa = w0 + l15;
  v8f acc[4];
  acc[0] = {}; acc[1] = {}; acc[2] = {}; acc[3] = {};
  const bfu16* brow = Wg + (size_t)(n0 + l15) * KCONV;   // + nt*16*KCONV per sub-tile
  for (int tap = 0; tap < 9; ++tap) {
    int ky = tap / 3, kx = tap - ky * 3;
    int hh = h + ky - 1;
    int ww = wa + kx - 1;
    bool inb = (hh >= 0) & (hh < HH) & (ww >= 0) & (ww < WWD);
    unsigned msk = inb ? 0xFFFFFFFFu : 0u;
    int hc = min(max(hh, 0), HH - 1);
    int wc = min(max(ww, 0), WWD - 1);
    const bfu16* ap = X + ((size_t)((b * HH + hc) * WWD + wc)) * CIN + half * 8;
    const bfu16* bp0 = brow + tap * CIN + half * 8;
    for (int c0 = 0; c0 < CIN; c0 += 32) {
      Frag16 af;
      af.u[0] = and4(*(const uint4*)(ap + c0), msk);       // unconditional global loads
      af.u[1] = and4(*(const uint4*)(ap + c0 + 16), msk);
      Frag16 bfr[4];
#pragma unroll
      for (int nt = 0; nt < 4; ++nt) {                     // stage all B fragments first
        const bfu16* bp = bp0 + (size_t)nt * 16 * KCONV + c0;
        bfr[nt].u[0] = *(const uint4*)(bp);
        bfr[nt].u[1] = *(const uint4*)(bp + 16);
      }
#pragma unroll
      for (int nt = 0; nt < 4; ++nt) {                     // then the WMMA burst
        acc[nt] = __builtin_amdgcn_wmma_f32_16x16x32_bf16(false, af.v, false, bfr[nt].v,
                                                          (short)0, acc[nt], false, false);
      }
    }
  }
#pragma unroll
  for (int nt = 0; nt < 4; ++nt) {
    int nn = n0 + nt * 16 + l15;
    float bv = bias[nn];
    for (int v = 0; v < 8; ++v) {
      int m = v + half * 8;
      float val = fmaxf(acc[nt][v] + bv, 0.f);   // relu
      size_t idx = (size_t)(m0 + m) * CMID + nn;
      RC[idx]  = val;
      RCb[idx] = f2bf(val);
    }
  }
}

// generic GEMM: C[M][N] = act(A[M][K] * B[N][K]^T + bias), bf16 in / f32 out.
// One wave per 16(M) x (16*NT)(N) strip; B fragments staged before WMMA burst.
template <int NT>
__global__ void gemm_wmma(const bfu16* __restrict__ A, const bfu16* __restrict__ B,
                          const float* __restrict__ bias, float* __restrict__ C,
                          int M, int N, int K, int relu) {
  int lane = threadIdx.x & 31;
  int half = lane >> 4, l15 = lane & 15;
  int n0 = blockIdx.x * (16 * NT);
  int m0 = blockIdx.y * 16;
  v8f acc[NT];
#pragma unroll
  for (int nt = 0; nt < NT; ++nt) acc[nt] = {};
  const bfu16* arow = A + (size_t)(m0 + l15) * K + half * 8;
  const bfu16* brow = B + (size_t)(n0 + l15) * K + half * 8;
  for (int k0 = 0; k0 < K; k0 += 32) {
    Frag16 af;
    af.u[0] = *(const uint4*)(arow + k0);
    af.u[1] = *(const uint4*)(arow + k0 + 16);
    if (k0 + 32 < K) __builtin_prefetch(arow + k0 + 32, 0, 1);   // global_prefetch_b8
    Frag16 bfr[NT];
#pragma unroll
    for (int nt = 0; nt < NT; ++nt) {
      const bfu16* bp = brow + (size_t)nt * 16 * K + k0;
      bfr[nt].u[0] = *(const uint4*)(bp);
      bfr[nt].u[1] = *(const uint4*)(bp + 16);
    }
#pragma unroll
    for (int nt = 0; nt < NT; ++nt) {
      acc[nt] = __builtin_amdgcn_wmma_f32_16x16x32_bf16(false, af.v, false, bfr[nt].v,
                                                        (short)0, acc[nt], false, false);
    }
  }
#pragma unroll
  for (int nt = 0; nt < NT; ++nt) {
    int nn = n0 + nt * 16 + l15;
    float bv = bias[nn];
    for (int v = 0; v < 8; ++v) {
      int m = v + half * 8;
      float val = acc[nt][v] + bv;
      if (relu) val = fmaxf(val, 0.f);
      C[(size_t)(m0 + m) * N + nn] = val;
    }
  }
}

// ---------------- gate epilogue ----------------

// alpha = sigmoid(dot(h1_row, gang_w2) + b2); RCg = bf16(RC * alpha)
__global__ void k_gate2(const float* __restrict__ H1, const float* __restrict__ w2,
                        const float* __restrict__ b2, const float* __restrict__ RC,
                        bfu16* __restrict__ RCg) {
  __shared__ float red[256];
  int row = blockIdx.x, tid = threadIdx.x;
  const float* h = H1 + (size_t)row * CMID;
  float p = 0.f;
  for (int j = tid; j < CMID; j += 256) p += h[j] * w2[j];
  red[tid] = p; __syncthreads();
  for (int s = 128; s > 0; s >>= 1) {
    if (tid < s) red[tid] += red[tid + s];
    __syncthreads();
  }
  float alpha = 1.f / (1.f + expf(-(red[0] + b2[0])));
  const float* r = RC + (size_t)row * CMID;
  bfu16* o = RCg + (size_t)row * CMID;
  for (int j = tid; j < CMID; j += 256) o[j] = f2bf(r[j] * alpha);
}

// ---------------- score + bbox decode ----------------

__global__ void k_epilogue(const float* __restrict__ OUTH, const float* __restrict__ im_info,
                           float* __restrict__ scores, float* __restrict__ boxes) {
  int n = blockIdx.x * blockDim.x + threadIdx.x;
  if (n >= MROWS) return;
  const float* r = OUTH + (size_t)n * NHEAD;
  float sproto = r[72];
  for (int j = 1; j < 20; ++j) sproto = fmaxf(sproto, r[72 + j]);
  float cls[18];
  for (int c = 0; c < 18; ++c) cls[c] = r[c] * (1.0f + r[18 + c]);   // cls*atten + cls
  for (int c = 1; c < 18; c += 2) cls[c] += 0.5f * sproto;           // LAMBDA_FG on odd ch
  int b = n >> 12;
  int hw = n & (HWSZ - 1);
  int h = hw >> 6, w = hw & 63;
  float imh = im_info[b * 3 + 0], imw = im_info[b * 3 + 1];
  float sx = (float)w * 16.f, sy = (float)h * 16.f;
  const float ratios[3] = {0.5f, 1.0f, 2.0f};
  for (int rr = 0; rr < 3; ++rr) {
    float wr = roundf(sqrtf(256.f / ratios[rr]));
    float hr = roundf(wr * ratios[rr]);
    for (int si = 0; si < 3; ++si) {
      int a = rr * 3 + si;
      float s  = (float)(8 << si);
      float wsz = wr * s, hsz = hr * s;
      float ax1 = 7.5f - 0.5f * (wsz - 1.f) + sx;
      float ay1 = 7.5f - 0.5f * (hsz - 1.f) + sy;
      float ax2 = 7.5f + 0.5f * (wsz - 1.f) + sx;
      float ay2 = 7.5f + 0.5f * (hsz - 1.f) + sy;
      float aw = ax2 - ax1 + 1.f, ah = ay2 - ay1 + 1.f;
      float acx = ax1 + 0.5f * aw, acy = ay1 + 0.5f * ah;
      const float* d = r + 36 + a * 4;
      float pcx = d[0] * aw + acx;
      float pcy = d[1] * ah + acy;
      float pw = expf(d[2]) * aw;
      float ph = expf(d[3]) * ah;
      float x1 = fminf(fmaxf(pcx - 0.5f * pw, 0.f), imw - 1.f);
      float y1 = fminf(fmaxf(pcy - 0.5f * ph, 0.f), imh - 1.f);
      float x2 = fminf(fmaxf(pcx + 0.5f * pw, 0.f), imw - 1.f);
      float y2 = fminf(fmaxf(pcy + 0.5f * ph, 0.f), imh - 1.f);
      int ai = hw * 9 + a;
      scores[(size_t)b * NANCH + ai] = 1.f / (1.f + expf(cls[a] - cls[9 + a]));
      float* bo = boxes + ((size_t)b * NANCH + ai) * 4;
      bo[0] = x1; bo[1] = y1; bo[2] = x2; bo[3] = y2;
    }
  }
}

// ---------------- top-6000 via histogram threshold ----------------

__global__ void k_hist(const float* __restrict__ scores, int* __restrict__ hist) {
  int i = blockIdx.x * blockDim.x + threadIdx.x;
  if (i >= NB * NANCH) return;
  int b = i / NANCH;
  int bin = (int)(scores[i] * (float)NBIN);
  bin = max(0, min(NBIN - 1, bin));
  atomicAdd(&hist[b * NBIN + bin], 1);
}

// Misc layout: [0..3]=tbin  [4..7]=above  [8..11]=rem  [12..15]=cntHi  [16..19]=cntT
__global__ void k_thresh(const int* __restrict__ hist, int* __restrict__ misc) {
  int b = blockIdx.x;
  if (threadIdx.x != 0) return;
  const int* hh = hist + b * NBIN;
  int acc = 0, t = 0, above = 0;
  for (int bin = NBIN - 1; bin >= 0; --bin) {
    if (acc + hh[bin] >= PRE) { t = bin; above = acc; break; }
    acc += hh[bin];
    if (bin == 0) { t = 0; above = acc; }
  }
  misc[b] = t; misc[4 + b] = above; misc[8 + b] = PRE - above;
}

__global__ void k_compact(const float* __restrict__ scores, const float* __restrict__ boxes,
                          int* __restrict__ misc,
                          float* __restrict__ candS, float* __restrict__ candB) {
  int i = blockIdx.x * blockDim.x + threadIdx.x;
  if (i >= NB * NANCH) return;
  int b = i / NANCH;
  float s = scores[i];
  int bin = (int)(s * (float)NBIN);
  bin = max(0, min(NBIN - 1, bin));
  int t = misc[b];
  int slot = -1;
  if (bin > t) {
    slot = atomicAdd(&misc[12 + b], 1);
  } else if (bin == t) {
    int q = atomicAdd(&misc[16 + b], 1);
    if (q < misc[8 + b]) slot = misc[4 + b] + q;
  }
  if (slot >= 0 && slot < PRE) {
    candS[(size_t)b * PRE + slot] = s;
    float* d = candB + ((size_t)b * PRE + slot) * 4;
    const float* sb = boxes + (size_t)i * 4;
    d[0] = sb[0]; d[1] = sb[1]; d[2] = sb[2]; d[3] = sb[3];
  }
}

// ---------------- NMS: one workgroup per batch ----------------

__global__ void k_nms(const float* __restrict__ candS, const float* __restrict__ candB,
                      float* __restrict__ out) {
  int b = blockIdx.x, tid = threadIdx.x;
  __shared__ float ssc[PRE];
  __shared__ float rv[256];
  __shared__ int   ri[256];
  __shared__ float sb[4];
  __shared__ int   sidx;
  for (int j = tid; j < PRE; j += 256) ssc[j] = candS[(size_t)b * PRE + j];
  __syncthreads();
  const float* bx = candB + (size_t)b * PRE * 4;
  for (int it = 0; it < POST; ++it) {
    float best = -1.f; int bi = -1;
    for (int j = tid; j < PRE; j += 256) {
      float v = ssc[j];
      if (v > best) { best = v; bi = j; }
    }
    rv[tid] = best; ri[tid] = bi; __syncthreads();
    for (int s = 128; s > 0; s >>= 1) {
      if (tid < s) {
        bool take = (rv[tid + s] > rv[tid]) ||
                    (rv[tid + s] == rv[tid] && ri[tid + s] >= 0 &&
                     (ri[tid] < 0 || ri[tid + s] < ri[tid]));
        if (take) { rv[tid] = rv[tid + s]; ri[tid] = ri[tid + s]; }
      }
      __syncthreads();
    }
    if (tid == 0) {
      float v = rv[0]; int k = ri[0];
      float* o = out + ((size_t)b * POST + it) * 5;
      o[0] = (float)b;
      if (v > 0.f && k >= 0) {
        sidx = k;
        sb[0] = bx[k * 4 + 0]; sb[1] = bx[k * 4 + 1];
        sb[2] = bx[k * 4 + 2]; sb[3] = bx[k * 4 + 3];
        o[1] = sb[0]; o[2] = sb[1]; o[3] = sb[2]; o[4] = sb[3];
      } else {
        sidx = -1;
        o[1] = 0.f; o[2] = 0.f; o[3] = 0.f; o[4] = 0.f;
      }
    }
    __syncthreads();
    if (sidx >= 0) {
      float x1 = sb[0], y1 = sb[1], x2 = sb[2], y2 = sb[3];
      float areaS = (x2 - x1 + 1.f) * (y2 - y1 + 1.f);
      for (int j = tid; j < PRE; j += 256) {
        if (ssc[j] < 0.f && j != sidx) continue;
        float bx1 = bx[j * 4], by1 = bx[j * 4 + 1], bx2 = bx[j * 4 + 2], by2 = bx[j * 4 + 3];
        float xx1 = fmaxf(x1, bx1), yy1 = fmaxf(y1, by1);
        float xx2 = fminf(x2, bx2), yy2 = fminf(y2, by2);
        float inter = fmaxf(0.f, xx2 - xx1 + 1.f) * fmaxf(0.f, yy2 - yy1 + 1.f);
        float areaJ = (bx2 - bx1 + 1.f) * (by2 - by1 + 1.f);
        float iou = inter / (areaS + areaJ - inter);
        if (iou > 0.7f) ssc[j] = -1.f;
      }
    }
    __syncthreads();
  }
}

// ---------------- launcher ----------------

extern "C" void kernel_launch(void* const* d_in, const int* in_sizes, int n_in,
                              void* d_out, int out_size, void* d_ws, size_t ws_size,
                              hipStream_t stream) {
  (void)in_sizes; (void)n_in; (void)out_size; (void)ws_size;
  const float* base_feat  = (const float*)d_in[0];
  const float* im_info    = (const float*)d_in[1];
  const float* attentions = (const float*)d_in[2];
  const float* P_ang      = (const float*)d_in[3];
  const float* P_pure     = (const float*)d_in[4];
  const float* conv_w     = (const float*)d_in[5];
  const float* conv_b     = (const float*)d_in[6];
  const float* cls_w      = (const float*)d_in[7];
  const float* cls_b      = (const float*)d_in[8];
  const float* bbox_w     = (const float*)d_in[9];
  const float* bbox_b     = (const float*)d_in[10];
  const float* merge_w    = (const float*)d_in[11];
  const float* merge_b    = (const float*)d_in[12];
  const float* meta_w1    = (const float*)d_in[13];
  const float* meta_b1    = (const float*)d_in[14];
  const float* meta_w2    = (const float*)d_in[15];
  const float* meta_b2    = (const float*)d_in[16];
  const float* meta_w3    = (const float*)d_in[17];
  const float* meta_b3    = (const float*)d_in[18];
  const float* proto_w    = (const float*)d_in[19];
  const float* proto_b    = (const float*)d_in[20];
  const float* gang_w1    = (const float*)d_in[21];
  const float* gang_b1    = (const float*)d_in[22];
  const float* gang_w2    = (const float*)d_in[23];
  const float* gang_b2    = (const float*)d_in[24];

  char* ws = (char*)d_ws;
  size_t off = 0;
  auto alloc = [&](size_t bytes) {
    size_t o = off; off += (bytes + 255) & ~(size_t)255; return o;
  };
  bfu16* Xb    = (bfu16*)(ws + alloc((size_t)MROWS * CIN * 2));   // 32MB NHWC bf16
  float* H1    = (float*)Xb;                                      // alias: gate1 out (X dead)
  bfu16* Wg    = (bfu16*)(ws + alloc((size_t)CMID * KCONV * 2));
  float* RC    = (float*)(ws + alloc((size_t)MROWS * CMID * 4));
  bfu16* RCb   = (bfu16*)(ws + alloc((size_t)MROWS * CMID * 2));
  bfu16* RCg   = (bfu16*)(ws + alloc((size_t)MROWS * CMID * 2));
  bfu16* W1g   = (bfu16*)(ws + alloc((size_t)CMID * CMID * 2));
  bfu16* Whead = (bfu16*)(ws + alloc((size_t)NHEAD * CMID * 2));
  float* BiasH = (float*)(ws + alloc(NHEAD * 4));
  float* OUTH  = (float*)(ws + alloc((size_t)MROWS * NHEAD * 4));
  float* Scor  = (float*)(ws + alloc((size_t)NB * NANCH * 4));
  float* Boxes = (float*)(ws + alloc((size_t)NB * NANCH * 16));
  float* CandS = (float*)(ws + alloc((size_t)NB * PRE * 4));
  float* CandB = (float*)(ws + alloc((size_t)NB * PRE * 16));
  int*   Hist  = (int*)  (ws + alloc((size_t)NB * NBIN * 4));
  int*   Misc  = (int*)  (ws + alloc(64 * 4));
  float* Pep   = (float*)(ws + alloc(64));
  float* AttS  = (float*)(ws + alloc(2048 * 4));
  float* GBias = (float*)(ws + alloc(512 * 4));
  float* Merg  = (float*)(ws + alloc(1024 * 4));
  float* H1m   = (float*)(ws + alloc(512 * 4));
  float* H2m   = (float*)(ws + alloc(128 * 4));
  float* CProt = (float*)(ws + alloc(32 * 4));

  // zero counters/histograms every call (determinism across graph replays)
  k_zero_int<<<(NB * NBIN + 255) / 256, 256, 0, stream>>>(Hist, NB * NBIN);
  k_zero_int<<<1, 64, 0, stream>>>(Misc, 64);

  // conversions
  k_cvt_x<<<(MROWS * CIN + 255) / 256, 256, 0, stream>>>(base_feat, Xb);
  k_cvt_wconv<<<(CMID * KCONV + 255) / 256, 256, 0, stream>>>(conv_w, Wg);
  k_cvt_w1g<<<(CMID * CMID + 255) / 256, 256, 0, stream>>>(gang_w1, W1g);

  // small meta chain (scalar-heavy but tiny)
  k_pep_attsum<<<1, 256, 0, stream>>>(P_ang, attentions, Pep, AttS);
  k_gate_bias<<<2, 256, 0, stream>>>(gang_w1, gang_b1, Pep, GBias);
  k_vecmat<<<4, 256, 0, stream>>>(AttS, merge_w, merge_b, Merg, 1024, 2048, 2);
  k_vecmat<<<2, 256, 0, stream>>>(Merg, meta_w1, meta_b1, H1m, 512, 1024, 1);
  k_vecmat<<<1, 256, 0, stream>>>(H1m, meta_w2, meta_b2, H2m, 128, 512, 1);
  k_dyn<<<(18 * CMID + 255) / 256, 256, 0, stream>>>(H2m, meta_w3, meta_b3, Whead);
  k_cproto<<<1, 32, 0, stream>>>(proto_b, P_pure, CProt);
  k_w2proto<<<(20 * CMID + 255) / 256, 256, 0, stream>>>(proto_w, P_pure, Whead);
  k_whead_static<<<(NHEAD * CMID + 255) / 256, 256, 0, stream>>>(
      cls_w, bbox_w, cls_b, bbox_b, CProt, Whead, BiasH);

  // heavy WMMA work
  conv3_wmma<<<dim3(CMID / 64, MROWS / 16), 32, 0, stream>>>(Xb, Wg, conv_b, RC, RCb);
  gemm_wmma<4><<<dim3(CMID / 64, MROWS / 16), 32, 0, stream>>>(
      RCb, W1g, GBias, H1, MROWS, CMID, CMID, 1);
  k_gate2<<<MROWS, 256, 0, stream>>>(H1, gang_w2, gang_b2, RC, RCg);
  gemm_wmma<2><<<dim3(NHEAD / 32, MROWS / 16), 32, 0, stream>>>(
      RCg, Whead, BiasH, OUTH, MROWS, NHEAD, CMID, 0);

  // proposal layer
  k_epilogue<<<(MROWS + 255) / 256, 256, 0, stream>>>(OUTH, im_info, Scor, Boxes);
  k_hist<<<(NB * NANCH + 255) / 256, 256, 0, stream>>>(Scor, Hist);
  k_thresh<<<NB, 32, 0, stream>>>(Hist, Misc);
  k_compact<<<(NB * NANCH + 255) / 256, 256, 0, stream>>>(Scor, Boxes, Misc, CandS, CandB);
  k_nms<<<NB, 256, 0, stream>>>(CandS, CandB, (float*)d_out);
}
